// Attention_43843026158375
// MI455X (gfx1250) — matrix operand order repaired and checked by reading, not compiled, not run on hardware
//
#include <hip/hip_runtime.h>
#include <hip/hip_bf16.h>

// ---------------------------------------------------------------------------
// Attention block for MI455X (gfx1250, wave32, WMMA f16 16x16x32, f32 acc)
// B=8, C=256, H=W=32, N=1024, HEADS=8, DH=64, INNER=512
// Register-blocked: conv waves do 2 output tiles, PV waves do 4.
// ---------------------------------------------------------------------------

typedef _Float16 h16;
typedef __attribute__((ext_vector_type(16))) _Float16 v16h;
typedef __attribute__((ext_vector_type(8)))  _Float16 v8h;
typedef __attribute__((ext_vector_type(8)))  float    v8f;
typedef __attribute__((ext_vector_type(4)))  float    v4f;

#define Bb     8
#define Cc     256
#define Hh     32
#define Nn     1024
#define HEADS  8
#define DH     64
#define INNER  512
#define PADW   34           // 32 + 1px halo each side
#define PPOS   (PADW*PADW)  // 1156

// ---------------- WMMA fragment helpers (ISA 7.12.2 layouts) ----------------

// A fragment: M rows in lanes (lane&15); halfs 0-15 -> K=16*(l>>4)..+15
// (contiguous, matches B fragments). M points at (row0*ldm + k0).
__device__ __forceinline__ v16h load_a_frag(const h16* M, int ldm, int lane) {
  int row = lane & 15, hs = lane >> 4;
  const h16* p = M + (size_t)row * ldm;
  v8h lo = *(const v8h*)(p + 16 * hs);
  v8h hi = *(const v8h*)(p + 16 * hs + 8);
  v16h r;
  #pragma unroll
  for (int i = 0; i < 8; ++i) { r[i] = lo[i]; r[i + 8] = hi[i]; }
  return r;
}

// B fragment from B^T stored row-major (N x K): lane = N col (lane&15),
// 16 contiguous K halfs at K = k0 + 16*(lane>>4). M points at (n0*ldm + k0).
__device__ __forceinline__ v16h load_b_frag(const h16* M, int ldm, int lane) {
  int col = lane & 15, hs = lane >> 4;
  return *(const v16h*)(M + (size_t)col * ldm + 16 * hs);
}

// Implicit-im2col B fragment (positions on N): inp_b = padded (34*34,256) f16.
// GEMM K index kk = tap*256 + c; a 32-wide k-step never crosses a tap.
__device__ __forceinline__ v16h load_b_conv(const h16* inp_b, int n0, int k0,
                                            int ntaps, int lane) {
  int tap = (ntaps == 1) ? 4 : (k0 >> 8);   // 1x1 conv == center tap (dy=dx=1)
  int dy = tap / 3, dx = tap % 3;
  int n  = n0 + (lane & 15);
  int py = (n >> 5) + dy, px = (n & 31) + dx;
  int c0 = (k0 & 255) + 16 * (lane >> 4);
  return *(const v16h*)(inp_b + (((size_t)py * PADW + px) << 8) + c0);
}

// Implicit-im2col A fragment (positions on M).
__device__ __forceinline__ v16h load_a_conv(const h16* inp_b, int n0, int k0,
                                            int lane) {
  int tap = k0 >> 8;
  int dy = tap / 3, dx = tap % 3;
  int n  = n0 + (lane & 15);
  int py = (n >> 5) + dy, px = (n & 31) + dx;
  int hs = lane >> 4;
  const h16* p = inp_b + (((size_t)py * PADW + px) << 8) + (k0 & 255);
  v8h lo = *(const v8h*)(p + 16 * hs);
  v8h hi = *(const v8h*)(p + 16 * hs + 8);
  v16h r;
  #pragma unroll
  for (int i = 0; i < 8; ++i) { r[i] = lo[i]; r[i + 8] = hi[i]; }
  return r;
}

__device__ __forceinline__ v8f wmma_f16(v16h a, v16h b, v8f c) {
  return __builtin_amdgcn_wmma_f32_16x16x32_f16(false, a, false, b,
                                                (short)0, c, false, false);
}

__device__ __forceinline__ v8h pack8(v8f acc) {
  v8h o;
  #pragma unroll
  for (int r = 0; r < 8; ++r) o[r] = (h16)acc[r];
  return o;
}

// ---------------- stage 1: inp = x + 2*sinu, padded, pos-major f16 ----------

__global__ void prep_inp(const float* __restrict__ x, h16* __restrict__ inp_t) {
  size_t id = (size_t)blockIdx.x * 256 + threadIdx.x;   // B*1156*256
  if (id >= (size_t)Bb * PPOS * Cc) return;
  int c   = id & 255;
  int pos = (int)((id >> 8) % PPOS);
  int b   = (int)(id / ((size_t)PPOS * Cc));
  int py = pos / PADW, px = pos % PADW;
  float v = 0.0f;
  if (py >= 1 && py <= 32 && px >= 1 && px <= 32) {
    int y = py - 1, xx = px - 1;
    int cc = c & 127;
    float arg;
    if (cc < 64) {
      float invf = __powf(10000.0f, -(2.0f * cc) / 128.0f);
      arg = (-1.0f + 2.0f * xx / 31.0f) * invf;       // width part
    } else {
      float invf = __powf(10000.0f, -(2.0f * (cc - 64)) / 128.0f);
      arg = (-1.0f + 2.0f * y / 31.0f) * invf;        // height part
    }
    float e = (c < 128) ? __sinf(arg) : __cosf(arg);
    v = x[(((size_t)b * Cc + c) * Hh + y) * Hh + xx] + 2.0f * e;
  }
  inp_t[((size_t)b * PPOS + pos) * Cc + c] = (h16)v;
}

// ---------------- stage 2: weight reorder/convert to f16 --------------------

__global__ void prep_weights(const float* __restrict__ wq, const float* __restrict__ wk,
                             const float* __restrict__ wv, const float* __restrict__ wo,
                             h16* __restrict__ wqA, h16* __restrict__ wkA,
                             h16* __restrict__ wvA, h16* __restrict__ woA) {
  size_t id = (size_t)blockIdx.x * 256 + threadIdx.x;   // 512*2304 max
  if (id < (size_t)INNER * 2304) {
    int oc = (int)(id / 2304);
    int kk = (int)(id % 2304);
    int tap = kk >> 8, c = kk & 255;                    // kk = tap*256 + c
    size_t s = (((size_t)oc * Cc) + c) * 9 + tap;       // OIHW flat
    wkA[id] = (h16)wk[s];
    wvA[id] = (h16)wv[s];
  }
  if (id < (size_t)INNER * Cc)    wqA[id] = (h16)wq[id];
  if (id < (size_t)INNER * INNER) woA[id] = (h16)wo[id];
}

// ---------------- stage 3a: Q/K convs: A = weights (oc=M), B = im2col -------
// Each wave computes an oc-tile PAIR sharing one im2col B fragment.
// D: lane = position, VGPR r walks d contiguously -> b128 stores.

__global__ void __launch_bounds__(256)
conv_qk(const h16* __restrict__ inp_t, const h16* __restrict__ wA,
        h16* __restrict__ dst, int ntaps) {
  int wave = blockIdx.x * 8 + (threadIdx.x >> 5);
  int lane = threadIdx.x & 31;
  int nt  = wave & 63;           // position tile
  int oc2 = (wave >> 6) & 15;    // 32-channel slab (pair of 16-wide oc tiles)
  int b   = wave >> 10;          // batch
  int Kdim = ntaps * Cc;
  const h16* inp_b = inp_t + (size_t)b * PPOS * Cc;
  const h16* w0 = wA + (size_t)(oc2 * 32) * Kdim;
  const h16* w1 = wA + (size_t)(oc2 * 32 + 16) * Kdim;
  v8f acc0 = {}, acc1 = {};
  for (int k0 = 0; k0 < Kdim; k0 += 32) {
    v16h bb = load_b_conv(inp_b, nt * 16, k0, ntaps, lane);
    v16h a0 = load_a_frag(w0 + k0, Kdim, lane);
    acc0 = wmma_f16(a0, bb, acc0);
    v16h a1 = load_a_frag(w1 + k0, Kdim, lane);
    acc1 = wmma_f16(a1, bb, acc1);
  }
  int hs = lane >> 4;
  int n = nt * 16 + (lane & 15);
  int ocb = oc2 * 32 + 8 * hs;            // 8 contiguous oc, never cross head
  int head = ocb >> 6, db = ocb & 63;
  *(v8h*)(dst + ((size_t)(b * HEADS + head) * Nn + n) * DH + db) = pack8(acc0);
  int ocb1 = ocb + 16;                    // stays inside the same 64-ch head
  int head1 = ocb1 >> 6, db1 = ocb1 & 63;
  *(v8h*)(dst + ((size_t)(b * HEADS + head1) * Nn + n) * DH + db1) = pack8(acc1);
}

// ---------------- stage 3b: V conv: A = im2col (pos=M), B = weights ---------
// Each wave computes an oc-tile PAIR sharing the im2col A fragment.
// D: lane = oc (-> d), VGPR r walks n contiguously -> b128 stores of V^T.

__global__ void __launch_bounds__(256)
conv_v(const h16* __restrict__ inp_t, const h16* __restrict__ wA,
       h16* __restrict__ dst) {
  int wave = blockIdx.x * 8 + (threadIdx.x >> 5);
  int lane = threadIdx.x & 31;
  int nt  = wave & 63;
  int oc2 = (wave >> 6) & 15;
  int b   = wave >> 10;
  const int Kdim = 9 * Cc;
  const h16* inp_b = inp_t + (size_t)b * PPOS * Cc;
  const h16* w0 = wA + (size_t)(oc2 * 32) * Kdim;
  const h16* w1 = wA + (size_t)(oc2 * 32 + 16) * Kdim;
  v8f acc0 = {}, acc1 = {};
  for (int k0 = 0; k0 < Kdim; k0 += 32) {
    v16h a  = load_a_conv(inp_b, nt * 16, k0, lane);
    v16h b0 = load_b_frag(w0 + k0, Kdim, lane);
    acc0 = wmma_f16(a, b0, acc0);
    v16h b1 = load_b_frag(w1 + k0, Kdim, lane);
    acc1 = wmma_f16(a, b1, acc1);
  }
  int hs = lane >> 4;
  int nb = nt * 16 + 8 * hs;              // 8 contiguous positions
  int oc0 = oc2 * 32 + (lane & 15);
  *(v8h*)(dst + ((size_t)(b * HEADS + (oc0 >> 6)) * DH + (oc0 & 63)) * Nn + nb)
      = pack8(acc0);
  int oc1 = oc0 + 16;
  *(v8h*)(dst + ((size_t)(b * HEADS + (oc1 >> 6)) * DH + (oc1 & 63)) * Nn + nb)
      = pack8(acc1);
}

// ---------------- stage 4: rotary embedding on q and k ----------------------

__global__ void rotary(h16* __restrict__ t) {
  size_t id = (size_t)blockIdx.x * 256 + threadIdx.x;   // B*HEADS*N*32
  if (id >= (size_t)Bb * HEADS * Nn * 32) return;
  int tt = id & 31;                 // pair index (d/2)
  int n  = (int)((id >> 5) & 1023);
  size_t bh = id >> 15;
  int a = n >> 5, bcol = n & 31;
  float scale = __expf((float)(tt & 15) * 0.10729586082f);  // 5^(t/15)
  int idx = (tt < 16) ? a : bcol;
  float f = (-1.0f + 2.0f * idx / 31.0f) * scale * 3.14159265358979f;
  float s = __sinf(f), c = __cosf(f);
  h16* p = t + ((size_t)bh * Nn + n) * DH + 2 * tt;
  float q0 = (float)p[0], q1 = (float)p[1];
  p[0] = (h16)(q0 * c - q1 * s);
  p[1] = (h16)(q1 * c + q0 * s);
}

// ---------------- stage 5: scores (keys=M, queries=N) + softmax -> P f16 ----
// One 256-thread WG per (b,h, 16-query strip). 16x1024 f32 strip in LDS.
// D: lane = query, VGPR r walks keys contiguously -> ds_store_b128 x2.

__global__ void __launch_bounds__(256)
scores_softmax(const h16* __restrict__ q, const h16* __restrict__ k,
               h16* __restrict__ P) {
  __shared__ float S[16 * 1024];            // 64 KB
  int wv = threadIdx.x >> 5, lane = threadIdx.x & 31;
  int hs = lane >> 4;
  int qt = blockIdx.x & 63;
  int bh = blockIdx.x >> 6;
  const h16* qb = q + ((size_t)bh * Nn + qt * 16) * DH;
  const h16* kb = k + (size_t)bh * Nn * DH;
  v16h b0 = load_b_frag(qb + 0, DH, lane);      // queries on N
  v16h b1 = load_b_frag(qb + 32, DH, lane);
  for (int t = 0; t < 8; ++t) {
    int kt = wv * 8 + t;                        // key tile for this wave
    v8f acc = {};
    v16h a0 = load_a_frag(kb + (size_t)kt * 16 * DH + 0, DH, lane);
    acc = wmma_f16(a0, b0, acc);
    v16h a1 = load_a_frag(kb + (size_t)kt * 16 * DH + 32, DH, lane);
    acc = wmma_f16(a1, b1, acc);
    float* srow = S + (lane & 15) * 1024 + kt * 16 + 8 * hs;
    *(v4f*)(srow)     = (v4f){acc[0], acc[1], acc[2], acc[3]};
    *(v4f*)(srow + 4) = (v4f){acc[4], acc[5], acc[6], acc[7]};
  }
  __syncthreads();
  const float SC = 0.125f;                      // 64^-0.5
  for (int rr = 0; rr < 2; ++rr) {
    int row = wv * 2 + rr;                      // 8 waves x 2 = 16 rows
    float* Srow = S + row * 1024 + lane * 32;   // 32 contiguous elems per lane
    float m = -3.4e38f;
    #pragma unroll
    for (int i = 0; i < 32; ++i) m = fmaxf(m, Srow[i]);
    #pragma unroll
    for (int off = 16; off >= 1; off >>= 1) m = fmaxf(m, __shfl_xor(m, off));
    float sum = 0.0f;
    #pragma unroll
    for (int i = 0; i < 32; ++i) {
      float e = __expf((Srow[i] - m) * SC);
      Srow[i] = e;
      sum += e;
    }
    #pragma unroll
    for (int off = 16; off >= 1; off >>= 1) sum += __shfl_xor(sum, off);
    float inv = 1.0f / sum;
    h16* Prow = P + ((size_t)bh * Nn + qt * 16 + row) * Nn + lane * 32;
    #pragma unroll
    for (int cch = 0; cch < 4; ++cch) {
      v8h o;
      #pragma unroll
      for (int i = 0; i < 8; ++i) o[i] = (h16)(Srow[cch * 8 + i] * inv);
      *(v8h*)(Prow + cch * 8) = o;              // b128 stores of P
    }
  }
}

// ---------------- stage 6: O^T = (P @ V)^T ----------------------------------
// Each wave computes ALL FOUR d-tiles of one query tile, reusing the P
// A-fragment 4x. D: lane = d, VGPR r walks queries -> float4 stores of O^T.

__global__ void __launch_bounds__(256)
pv_gemm(const h16* __restrict__ P, const h16* __restrict__ vT,
        float* __restrict__ O) {
  int wave = blockIdx.x * 8 + (threadIdx.x >> 5);
  int lane = threadIdx.x & 31;
  int mt = wave & 63;            // query tile
  int bh = wave >> 6;            // (b*HEADS + h)
  const h16* Pb = P  + ((size_t)bh * Nn + mt * 16) * Nn;
  const h16* vb = vT + (size_t)bh * DH * Nn;
  v8f acc0 = {}, acc1 = {}, acc2 = {}, acc3 = {};
  for (int kc = 0; kc < 32; ++kc) {
    int k0 = kc * 32;
    v16h a = load_a_frag(Pb + k0, Nn, lane);
    if (kc + 1 < 32)                                    // CDNA5 prefetch path
      __builtin_prefetch(Pb + k0 + 32 + (size_t)(lane & 15) * Nn, 0, 1);
    v16h b0 = load_b_frag(vb + (size_t)0 * 16 * Nn + k0, Nn, lane);
    acc0 = wmma_f16(a, b0, acc0);
    v16h b1 = load_b_frag(vb + (size_t)1 * 16 * Nn + k0, Nn, lane);
    acc1 = wmma_f16(a, b1, acc1);
    v16h b2 = load_b_frag(vb + (size_t)2 * 16 * Nn + k0, Nn, lane);
    acc2 = wmma_f16(a, b2, acc2);
    v16h b3 = load_b_frag(vb + (size_t)3 * 16 * Nn + k0, Nn, lane);
    acc3 = wmma_f16(a, b3, acc3);
  }
  int hs = lane >> 4;
  int nb = mt * 16 + 8 * hs;
  #pragma unroll
  for (int dt = 0; dt < 4; ++dt) {
    v8f acc = (dt == 0) ? acc0 : (dt == 1) ? acc1 : (dt == 2) ? acc2 : acc3;
    int d = dt * 16 + (lane & 15);
    float* orow = O + ((size_t)bh * DH + d) * Nn + nb;
    *(v4f*)(orow)     = (v4f){acc[0], acc[1], acc[2], acc[3]};
    *(v4f*)(orow + 4) = (v4f){acc[4], acc[5], acc[6], acc[7]};
  }
}

// ---------------- stage 7: talking-heads mix -> (b, n, g*64+d) f16 ----------
// Reads O^T (b,h,d,n).

__global__ void mix_heads(const float* __restrict__ O, const float* __restrict__ mixw,
                          h16* __restrict__ mixed) {
  size_t id = (size_t)blockIdx.x * 256 + threadIdx.x;   // B*N*INNER
  if (id >= (size_t)Bb * Nn * INNER) return;
  int ch = id & 511;                  // g*64 + d
  int n  = (int)((id >> 9) & 1023);
  int b  = (int)(id >> 19);
  int g = ch >> 6, d = ch & 63;
  float s = 0.0f;
  #pragma unroll
  for (int h = 0; h < HEADS; ++h)
    s += mixw[h * HEADS + g] * O[(((size_t)b * HEADS + h) * DH + d) * Nn + n];
  mixed[((size_t)b * Nn + n) * INNER + ch] = (h16)s;
}

// ---------------- stage 8: 1x1 out conv + bias + GLU ------------------------
// A = mixed (pos=M), B = wo rows (oc=N). Paired tiles (oc, oc+256) share A.
// D: lane = oc, VGPR r walks n contiguously -> float4 output stores.

__global__ void __launch_bounds__(256)
out_glu(const h16* __restrict__ woA, const h16* __restrict__ mixed,
        const float* __restrict__ bo, float* __restrict__ out) {
  int wave = blockIdx.x * 8 + (threadIdx.x >> 5);
  int lane = threadIdx.x & 31;
  int nt = wave & 63;
  int mt = (wave >> 6) & 15;     // oc tile within first 256 channels
  int b  = wave >> 10;
  const h16* mb = mixed + ((size_t)b * Nn + nt * 16) * INNER;
  v8f aa = {}, gg = {};
  for (int kc = 0; kc < 16; ++kc) {
    int k0 = kc * 32;
    v16h a  = load_a_frag(mb + k0, INNER, lane);
    v16h wa = load_b_frag(woA + (size_t)(mt * 16) * INNER + k0, INNER, lane);
    aa = wmma_f16(a, wa, aa);
    v16h wg = load_b_frag(woA + (size_t)(mt * 16 + 256) * INNER + k0, INNER, lane);
    gg = wmma_f16(a, wg, gg);
  }
  int hs = lane >> 4;
  int oc = mt * 16 + (lane & 15);
  float ba = bo[oc], bg = bo[oc + 256];
  float r0[8];
  #pragma unroll
  for (int r = 0; r < 8; ++r) {
    float a = aa[r] + ba;
    float g = gg[r] + bg;
    r0[r] = a * (1.0f / (1.0f + __expf(-g)));
  }
  float* orow = out + ((size_t)b * Cc + oc) * Nn + nt * 16 + 8 * hs;
  *(v4f*)(orow)     = (v4f){r0[0], r0[1], r0[2], r0[3]};
  *(v4f*)(orow + 4) = (v4f){r0[4], r0[5], r0[6], r0[7]};
}

// ---------------------------------------------------------------------------

extern "C" void kernel_launch(void* const* d_in, const int* in_sizes, int n_in,
                              void* d_out, int out_size, void* d_ws, size_t ws_size,
                              hipStream_t stream) {
  const float* x    = (const float*)d_in[0];
  const float* wq   = (const float*)d_in[1];
  const float* wk   = (const float*)d_in[2];
  const float* wv   = (const float*)d_in[3];
  const float* mixw = (const float*)d_in[4];
  const float* wo   = (const float*)d_in[5];
  const float* bo   = (const float*)d_in[6];
  float* out = (float*)d_out;

  // workspace carve-up (256B aligned)
  char* base = (char*)d_ws;
  size_t off = 0;
  auto carve = [&](size_t bytes) -> char* {
    char* p = base + off;
    off += (bytes + 255) & ~(size_t)255;
    return p;
  };
  h16*   inp_t = (h16*)carve((size_t)Bb * PPOS * Cc * 2);          // 4.7 MB
  h16*   wqA   = (h16*)carve((size_t)INNER * Cc * 2);
  h16*   wkA   = (h16*)carve((size_t)INNER * 2304 * 2);
  h16*   wvA   = (h16*)carve((size_t)INNER * 2304 * 2);
  h16*   woA   = (h16*)carve((size_t)INNER * INNER * 2);
  h16*   qbuf  = (h16*)carve((size_t)Bb * HEADS * Nn * DH * 2);    // 8 MB
  h16*   kbuf  = (h16*)carve((size_t)Bb * HEADS * Nn * DH * 2);
  h16*   vTbuf = (h16*)carve((size_t)Bb * HEADS * Nn * DH * 2);
  h16*   Pbuf  = (h16*)carve((size_t)Bb * HEADS * Nn * Nn * 2);    // 134 MB
  float* Obuf  = (float*)carve((size_t)Bb * HEADS * Nn * DH * 4);  // 16 MB
  h16*   mbuf  = (h16*)carve((size_t)Bb * Nn * INNER * 2);         // 8 MB
  (void)ws_size; (void)in_sizes; (void)n_in; (void)out_size;

  prep_inp<<<9248, 256, 0, stream>>>(x, inp_t);
  prep_weights<<<4608, 256, 0, stream>>>(wq, wk, wv, wo, wqA, wkA, wvA, woA);

  conv_qk<<<1024, 256, 0, stream>>>(inp_t, wqA, qbuf, 1);   // Q  (1x1)
  conv_qk<<<1024, 256, 0, stream>>>(inp_t, wkA, kbuf, 9);   // K  (3x3)
  conv_v <<<1024, 256, 0, stream>>>(inp_t, wvA, vTbuf);     // V^T (3x3)

  rotary<<<8192, 256, 0, stream>>>(qbuf);
  rotary<<<8192, 256, 0, stream>>>(kbuf);

  scores_softmax<<<4096, 256, 0, stream>>>(qbuf, kbuf, Pbuf);
  pv_gemm<<<512, 256, 0, stream>>>(Pbuf, vTbuf, Obuf);
  mix_heads<<<16384, 256, 0, stream>>>(Obuf, mixw, mbuf);
  out_glu<<<1024, 256, 0, stream>>>(woA, mbuf, bo, out);
}
